// LatentPhaseStateReparameterizer_88948772700910
// MI455X (gfx1250) — compile-verified, matrix operand order checked by reading, not run on hardware
//
#include <hip/hip_runtime.h>

// ---------------------------------------------------------------------------
// LatentPhaseStateReparameterizer — MI455X (gfx1250) pipeline.
//  * CNN encoder: direct f32 conv + fused GroupNorm/GELU block reductions.
//  * MLP heads:   LDS-staged f16 WMMA GEMMs (V_WMMA_F32_16X16X32_F16),
//                 fully unrolled K-loop, 32x64 C-tile per 4-wave block.
//  * (B,T,T) cost/softmax/contract: fused row kernel; per-batch vectors
//    staged into LDS via the Tensor Data Mover (tensor_load_to_lds +
//    s_wait_tensorcnt); fixed-order reductions (bit-deterministic).
// ---------------------------------------------------------------------------

typedef __attribute__((ext_vector_type(16))) _Float16 v16h;
typedef __attribute__((ext_vector_type(8)))  _Float16 v8h;
typedef __attribute__((ext_vector_type(8)))  float    v8f;
typedef __attribute__((ext_vector_type(4)))  unsigned int v4u;
typedef __attribute__((ext_vector_type(8)))  int      v8i;
typedef __attribute__((ext_vector_type(4)))  int      v4i;

#define T_STEPS 1024
#define BATCH   32
#define INV1023 (1.0f / 1023.0f)

__device__ __forceinline__ float gelu_f(float x) {
    return 0.5f * x * (1.0f + erff(x * 0.7071067811865476f));
}
__device__ __forceinline__ float softplus_f(float x) {
    return (x > 20.0f) ? x : log1pf(__expf(x));
}
__device__ __forceinline__ float block_sum256(float v, float* red, int tid) {
    red[tid] = v; __syncthreads();
    for (int s = 128; s > 0; s >>= 1) {
        if (tid < s) red[tid] += red[tid + s];
        __syncthreads();
    }
    float r = red[0]; __syncthreads();
    return r;
}

// ---------------------------------------------------------------------------
// Tensor Data Mover: DMA `nelem` f32 from global -> LDS (ISA ch.8 D#).
// 2-D descriptor: tile_dim0 = nelem (<= 65535), tile_dim1 = 1. Wave-uniform
// arguments; issued once per wave (EXEC ignored for TENSOR ops).
// ---------------------------------------------------------------------------
__device__ __forceinline__ void tdm_load_f32(const float* g, float* l,
                                             unsigned int nelem) {
    unsigned long long ga = (unsigned long long)(uintptr_t)g;
    unsigned int lds = (unsigned int)(uintptr_t)l;   // low 32 bits = LDS offset
    v4u g0;
    g0[0] = 1u;                                          // count=1, user mode
    g0[1] = lds;                                         // lds_addr (bytes)
    g0[2] = (unsigned int)(ga & 0xffffffffull);          // global_addr[31:0]
    g0[3] = (unsigned int)((ga >> 32) & 0x01ffffffull)   // global_addr[56:32]
          | (2u << 30);                                  // type = 2 ("image")
    v8i g1;
    g1[0] = (int)(2u << 16);                             // data_size = 4B
    g1[1] = (int)(nelem << 16);                          // tensor_dim0[15:0]
    g1[2] = (int)((nelem >> 16) | (1u << 16));           // dim0 hi | tensor_dim1=1
    g1[3] = (int)(nelem << 16);                          // tile_dim0
    g1[4] = 1;                                           // tile_dim1 = 1
    g1[5] = (int)nelem;                                  // tensor_dim0_stride lo
    g1[6] = (int)((nelem & 0xffffu) << 16);              // dim0_stride hi=0 | dim1_stride lo
    g1[7] = 0;
    v4i z4 = {0, 0, 0, 0};
#if defined(__clang_major__) && (__clang_major__ >= 23)
    v8i z8 = {0, 0, 0, 0, 0, 0, 0, 0};
    __builtin_amdgcn_tensor_load_to_lds(g0, g1, z4, z4, z8, 0);
#else
    __builtin_amdgcn_tensor_load_to_lds(g0, g1, z4, z4, 0);
#endif
}

// ---------------------------------------------------------------------------
// Direct 3x3 conv, pad=1, arbitrary stride. f32. One output element / thread.
// ---------------------------------------------------------------------------
__global__ void conv3x3_k(const float* __restrict__ x, const float* __restrict__ w,
                          const float* __restrict__ bias, float* __restrict__ y,
                          int Cin, int Cout, int Hin, int Win,
                          int stride, int Hout, int Wout) {
    int idx = blockIdx.x * blockDim.x + threadIdx.x;
    int total = BATCH * Cout * Hout * Wout;
    if (idx >= total) return;
    int wo = idx % Wout; int t = idx / Wout;
    int ho = t % Hout;   t /= Hout;
    int co = t % Cout;   int b = t / Cout;
    float acc = bias[co];
    int hb = ho * stride - 1, wb = wo * stride - 1;
    for (int ci = 0; ci < Cin; ++ci) {
        const float* xp = x + ((size_t)(b * Cin + ci) * Hin) * Win;
        const float* wp = w + (size_t)(co * Cin + ci) * 9;
        #pragma unroll
        for (int kh = 0; kh < 3; ++kh) {
            int h = hb + kh;
            if (h < 0 || h >= Hin) continue;
            #pragma unroll
            for (int kw = 0; kw < 3; ++kw) {
                int ww = wb + kw;
                if (ww < 0 || ww >= Win) continue;
                acc += xp[h * Win + ww] * wp[kh * 3 + kw];
            }
        }
    }
    y[idx] = acc;
}

// ---------------------------------------------------------------------------
// GroupNorm(groups=4) + exact GELU. One block per (batch, group).
// ---------------------------------------------------------------------------
__global__ void gn_gelu_k(const float* __restrict__ x, const float* __restrict__ gamma,
                          const float* __restrict__ beta, float* __restrict__ y,
                          int C, int HW) {
    __shared__ float r1[256], r2[256];
    int g = blockIdx.x & 3, b = blockIdx.x >> 2;
    int Cg = C >> 2;
    int n = Cg * HW;
    const float* xp = x + ((size_t)b * C + g * Cg) * HW;
    float s = 0.0f, s2 = 0.0f;
    for (int k = threadIdx.x; k < n; k += 256) { float v = xp[k]; s += v; s2 += v * v; }
    r1[threadIdx.x] = s; r2[threadIdx.x] = s2; __syncthreads();
    for (int st = 128; st > 0; st >>= 1) {
        if (threadIdx.x < st) { r1[threadIdx.x] += r1[threadIdx.x + st]; r2[threadIdx.x] += r2[threadIdx.x + st]; }
        __syncthreads();
    }
    float inv_n = 1.0f / (float)n;
    float mean = r1[0] * inv_n;
    float var  = r2[0] * inv_n - mean * mean;
    float rstd = rsqrtf(var + 1e-5f);
    float* yp = y + ((size_t)b * C + g * Cg) * HW;
    for (int k = threadIdx.x; k < n; k += 256) {
        int c = g * Cg + k / HW;
        float v = (xp[k] - mean) * rstd * gamma[c] + beta[c];
        yp[k] = gelu_f(v);
    }
}

// ---------------------------------------------------------------------------
// Global average pool over 16x16. One output channel per thread.
// ---------------------------------------------------------------------------
__global__ void avgpool_k(const float* __restrict__ x, float* __restrict__ z) {
    int idx = blockIdx.x * blockDim.x + threadIdx.x;
    if (idx >= BATCH * 128) return;
    const float* p = x + (size_t)idx * 256;
    float s = 0.0f;
    for (int k = 0; k < 256; ++k) s += p[k];
    z[idx] = s * (1.0f / 256.0f);
}

// ---------------------------------------------------------------------------
// WMMA GEMM, M = 32 fixed, K/activation compile-time:
//   C[32,N] = act( A[32,K] @ W[K,N] + bias )
// Block = 128 threads (4 waves), computes a 32x64 C-tile.
// A staged to LDS as f16 [32][K], W staged TRANSPOSED as f16 [64][K].
// Per k-step each lane builds its v16h operands with two aligned 16-byte LDS
// reads (ISA 7.12.2: halves h<8 -> K = k0+8*(lane>>4)+h, h>=8 -> +16).
// ACT: 0=none, 1=gelu, 2=sigmoid.
// ---------------------------------------------------------------------------
#define KPAD 136   // 128 + 8 halves, row stride = 272B (16B aligned)

template <int K, int ACT>
__global__ void gemm_wmma_k(const float* __restrict__ A, const float* __restrict__ W,
                            const float* __restrict__ bias, float* __restrict__ C,
                            int N) {
    __shared__ __align__(16) _Float16 sA[32][KPAD];
    __shared__ __align__(16) _Float16 sW[64][KPAD];
    int tid = threadIdx.x;
    int n0 = blockIdx.x * 64;

    // stage A (32 x K), coalesced along K
    for (int idx = tid; idx < 32 * K; idx += 128) {
        int r = idx / K, k = idx % K;
        sA[r][k] = (_Float16)A[r * K + k];
    }
    // stage W transposed (64 cols x K), coalesced along columns
    for (int idx = tid; idx < 64 * K; idx += 128) {
        int k = idx >> 6, c = idx & 63;
        int col = n0 + c;
        float v = (col < N) ? W[(size_t)k * N + col] : 0.0f;
        sW[c][k] = (_Float16)v;
    }
    __syncthreads();

    int wv  = tid >> 5;          // wave id 0..3 -> column sub-tile
    int lane = tid & 31;
    int l15 = lane & 15;
    int hi8 = (lane >> 4) << 3;
    const _Float16* ar0 = &sA[l15][0];
    const _Float16* ar1 = &sA[16 + l15][0];
    const _Float16* wr  = &sW[wv * 16 + l15][0];

    v8f acc0 = {}, acc1 = {};
    #pragma unroll
    for (int k0 = 0; k0 < K; k0 += 32) {
        v8h alo0 = *(const v8h*)(ar0 + k0 + hi8);
        v8h ahi0 = *(const v8h*)(ar0 + k0 + 16 + hi8);
        v8h alo1 = *(const v8h*)(ar1 + k0 + hi8);
        v8h ahi1 = *(const v8h*)(ar1 + k0 + 16 + hi8);
        v8h blo  = *(const v8h*)(wr  + k0 + hi8);
        v8h bhi  = *(const v8h*)(wr  + k0 + 16 + hi8);
        v16h a0 = __builtin_shufflevector(alo0, ahi0, 0,1,2,3,4,5,6,7,8,9,10,11,12,13,14,15);
        v16h a1 = __builtin_shufflevector(alo1, ahi1, 0,1,2,3,4,5,6,7,8,9,10,11,12,13,14,15);
        v16h bm = __builtin_shufflevector(blo,  bhi,  0,1,2,3,4,5,6,7,8,9,10,11,12,13,14,15);
        acc0 = __builtin_amdgcn_wmma_f32_16x16x32_f16(false, a0, false, bm,
                                                      (short)0, acc0, false, false);
        acc1 = __builtin_amdgcn_wmma_f32_16x16x32_f16(false, a1, false, bm,
                                                      (short)0, acc1, false, false);
    }

    int col = n0 + wv * 16 + l15;
    if (col < N) {
        float bb = bias[col];
        #pragma unroll
        for (int r = 0; r < 8; ++r) {
            float v0 = acc0[r] + bb;
            float v1 = acc1[r] + bb;
            if (ACT == 1)      { v0 = gelu_f(v0); v1 = gelu_f(v1); }
            else if (ACT == 2) { v0 = 1.0f / (1.0f + __expf(-v0));
                                 v1 = 1.0f / (1.0f + __expf(-v1)); }
            int row0 = r + hi8;
            int row1 = 16 + r + hi8;
            C[(size_t)row0 * N + col] = v0;
            C[(size_t)row1 * N + col] = v1;
        }
    }
}

// ---------------------------------------------------------------------------
// monotone(): softplus cumsum, normalized, leading 0 (1-x if invert).
// ---------------------------------------------------------------------------
__global__ void monotone_k(const float* __restrict__ logits, float* __restrict__ out,
                           int invert) {
    int b = blockIdx.x * blockDim.x + threadIdx.x;
    if (b >= BATCH) return;
    const float* l = logits + (size_t)b * 1023;
    float cum = 0.0f;
    for (int k = 0; k < 1023; ++k) cum += softplus_f(l[k]) + 1e-6f;
    float inv = 1.0f / fmaxf(cum, 1e-6f);
    float* o = out + (size_t)b * T_STEPS;
    o[0] = invert ? 1.0f : 0.0f;
    float run = 0.0f;
    for (int k = 0; k < 1023; ++k) {
        run += softplus_f(l[k]) + 1e-6f;
        float p = run * inv;
        o[k + 1] = invert ? (1.0f - p) : p;
    }
}

// ---------------------------------------------------------------------------
// Fused assignment row kernel: one 256-thread block per (b, i) row.
// Per-batch vectors DMA'd into LDS by the Tensor Data Mover (wave 0),
// then: cost -> (min,argmin) -> stable softmax -> write assignment row ->
// 4 contractions + confidence, all fixed-order LDS reductions.
// LI=1, LP=1, LE=0.5, TEMP=0.25, REF_P=1.
// ---------------------------------------------------------------------------
__global__ void assign_rows_k(const float* __restrict__ pp_all,
                              const float* __restrict__ le_all,
                              const float* __restrict__ mis_all,
                              const float* __restrict__ alpha,
                              float* __restrict__ assign_out,
                              float* __restrict__ spi, float* __restrict__ sp,
                              float* __restrict__ se,  float* __restrict__ sa,
                              float* __restrict__ confRow, int* __restrict__ amin) {
    __shared__ __align__(16) float s_le[T_STEPS];
    __shared__ __align__(16) float s_al[T_STEPS];
    __shared__ float redv[256];
    __shared__ int   redi[256];
    int tid = threadIdx.x;
    int b = blockIdx.x >> 10;
    int i = blockIdx.x & 1023;
    const float* leb = le_all + (size_t)b * T_STEPS;
    // Tensor Data Mover: 2x 4KB global -> LDS, issued once by wave 0.
    if (tid < 32) {
        tdm_load_f32(leb,   s_le, T_STEPS);
        tdm_load_f32(alpha, s_al, T_STEPS);
        __builtin_amdgcn_s_wait_tensorcnt(0);
    }
    __syncthreads();

    float gi   = i * INV1023;
    float ppv  = pp_all[(size_t)b * T_STEPS + i];
    float mv   = mis_all[(size_t)b * T_STEPS + i];
    float refi = 1.0f - gi;               // REF_P = 1
    float wI   = 1.0f * (1.0f - mv);      // LI
    float wE   = 0.5f * (1.0f + mv);      // LE

    float cst[4];
    float bestv = 3.0e38f; int besti = 0x7fffffff;
    #pragma unroll
    for (int k = 0; k < 4; ++k) {
        int j = (k << 8) + tid;
        float gj = j * INV1023;
        float dI = gj - gi, dP = gj - ppv, dE = s_le[j] - refi;
        float c = wI * dI * dI + dP * dP + wE * dE * dE;   // LP = 1
        cst[k] = c;
        if (c < bestv || (c == bestv && j < besti)) { bestv = c; besti = j; }
    }
    // (min, first-index) block reduction
    redv[tid] = bestv; redi[tid] = besti; __syncthreads();
    for (int s = 128; s > 0; s >>= 1) {
        if (tid < s) {
            float ov = redv[tid + s]; int oi = redi[tid + s];
            if (ov < redv[tid] || (ov == redv[tid] && oi < redi[tid])) {
                redv[tid] = ov; redi[tid] = oi;
            }
        }
        __syncthreads();
    }
    float cmin = redv[0]; int argmin_j = redi[0];
    __syncthreads();

    // stable softmax(-cost/TEMP): exp((cmin - c) * 4)
    float e[4]; float lsum = 0.0f;
    #pragma unroll
    for (int k = 0; k < 4; ++k) { e[k] = __expf((cmin - cst[k]) * 4.0f); lsum += e[k]; }
    float total = block_sum256(lsum, redv, tid);
    float inv_sum = 1.0f / total;

    float aspi = 0.0f, asp = 0.0f, ase = 0.0f, asa = 0.0f, acf = 0.0f;
    float* outrow = assign_out + ((size_t)(b * T_STEPS + i)) * T_STEPS;
    #pragma unroll
    for (int k = 0; k < 4; ++k) {
        int j = (k << 8) + tid;
        float a_ = e[k] * inv_sum;
        outrow[j] = a_;
        aspi += a_ * (float)j;
        asp  += a_ * (j * INV1023);
        ase  += a_ * s_le[j];
        asa  += a_ * s_al[j];
        acf  += a_ * (1.0f - a_);
    }
    aspi = block_sum256(aspi, redv, tid);
    asp  = block_sum256(asp,  redv, tid);
    ase  = block_sum256(ase,  redv, tid);
    asa  = block_sum256(asa,  redv, tid);
    acf  = block_sum256(acf,  redv, tid);
    if (tid == 0) {
        size_t r = (size_t)b * T_STEPS + i;
        spi[r] = aspi; sp[r] = asp; se[r] = ase; sa[r] = asa;
        confRow[r] = acf; amin[r] = argmin_j;
    }
}

// ---------------------------------------------------------------------------
// cummax along i per batch, with [:,0]=0 and [:,T-1]=T-1 overrides.
// ---------------------------------------------------------------------------
__global__ void hard_cummax_k(const int* __restrict__ amin, int* __restrict__ hard) {
    int b = threadIdx.x;
    if (b >= BATCH) return;
    int m = -2147483647;
    for (int i = 0; i < T_STEPS; ++i) {
        int v = amin[(size_t)b * T_STEPS + i];
        m = (v > m) ? v : m;
        int h = m;
        if (i == 0) h = 0;
        if (i == T_STEPS - 1) h = T_STEPS - 1;
        hard[(size_t)b * T_STEPS + i] = h;
    }
}

// ---------------------------------------------------------------------------
// Gather final per-batch outputs at nominal indices.
// ---------------------------------------------------------------------------
__global__ void gather_out_k(const int* __restrict__ nom, const float* __restrict__ spi,
                             const float* __restrict__ sa, const int* __restrict__ hard,
                             float* __restrict__ out_phase, float* __restrict__ out_alpha,
                             float* __restrict__ out_hard) {
    int b = threadIdx.x;
    if (b >= BATCH) return;
    int t = nom[b];
    t = (t < 0) ? 0 : ((t > T_STEPS - 1) ? T_STEPS - 1 : t);
    out_phase[b] = spi[(size_t)b * T_STEPS + t];
    out_alpha[b] = sa[(size_t)b * T_STEPS + t];
    out_hard[b]  = (float)hard[(size_t)b * T_STEPS + t];
}

// ---------------------------------------------------------------------------
// phase_loss: deterministic single-block reduction over B*T rows.
// ---------------------------------------------------------------------------
__global__ void loss_k(const float* __restrict__ mis, const float* __restrict__ se,
                       const float* __restrict__ sp, const float* __restrict__ confR,
                       float* __restrict__ out_loss) {
    __shared__ float red[256];
    int tid = threadIdx.x;
    float sA = 0, sS = 0, sMo = 0, sI = 0, sM = 0, sC = 0;
    const float STEP = 1.0f / 1023.0f;
    for (int r = tid; r < BATCH * T_STEPS; r += 256) {
        int i = r & 1023;
        float gi = i * INV1023, refi = 1.0f - gi;
        float m = mis[r], sev = se[r], spv = sp[r];
        float dA = sev - refi; sA += (1.0f + m) * dA * dA;
        float dI = spv - gi;   sI += (1.0f - m) * dI * dI;
        sM += m; sC += confR[r];
        if (i < T_STEPS - 1) {
            float nxt = sp[r + 1];
            float d = nxt - spv - STEP; sS += d * d;
            float mg = fmaxf(spv - nxt, 0.0f); sMo += mg * mg;
        }
    }
    sA  = block_sum256(sA,  red, tid);
    sS  = block_sum256(sS,  red, tid);
    sMo = block_sum256(sMo, red, tid);
    sI  = block_sum256(sI,  red, tid);
    sM  = block_sum256(sM,  red, tid);
    sC  = block_sum256(sC,  red, tid);
    if (tid == 0) {
        float nBT = (float)(BATCH * T_STEPS);
        float nS  = (float)(BATCH * (T_STEPS - 1));
        float align  = sA / nBT;
        float smooth = sS / nS + 0.5f * (sMo / nS);
        float ident  = sI / nBT;
        float msl    = sM / nBT;
        float conf   = sC / (nBT * (float)T_STEPS);
        out_loss[0] = 1.0f * align + 0.1f * smooth + 0.05f * ident
                    + 0.001f * msl + 0.001f * conf;
    }
}

// ---------------------------------------------------------------------------
// Host-side launch. Input index map = jax pytree flatten of setup_inputs():
//   0 x_obs, 1 alpha_schedule, 2 nominal_index, then params (sorted keys):
//   blocks[0..2] x {b1,b2,bg1,bg2,g1,g2,w1,w2}, energy[(w,b)x3],
//   mismatch[(w,b)x2], phase[(w,b)x3], proj[(w,b)x2].
// ---------------------------------------------------------------------------
enum {
    IN_XOBS = 0, IN_ALPHA = 1, IN_NOM = 2,
    B0 = 3, B1 = 11, B2 = 19,         // each: b1,b2,bg1,bg2,g1,g2,w1,w2
    EN_W0 = 27, EN_B0, EN_W1, EN_B1, EN_W2, EN_B2,
    MM_W0 = 33, MM_B0, MM_W1, MM_B1,
    PH_W0 = 37, PH_B0, PH_W1, PH_B1, PH_W2, PH_B2,
    PR_W0 = 43, PR_B0, PR_W1, PR_B1
};
#define BLK_B1(base)  ((const float*)d_in[(base) + 0])
#define BLK_B2(base)  ((const float*)d_in[(base) + 1])
#define BLK_BG1(base) ((const float*)d_in[(base) + 2])
#define BLK_BG2(base) ((const float*)d_in[(base) + 3])
#define BLK_G1(base)  ((const float*)d_in[(base) + 4])
#define BLK_G2(base)  ((const float*)d_in[(base) + 5])
#define BLK_W1(base)  ((const float*)d_in[(base) + 6])
#define BLK_W2(base)  ((const float*)d_in[(base) + 7])

extern "C" void kernel_launch(void* const* d_in, const int* in_sizes, int n_in,
                              void* d_out, int out_size, void* d_ws, size_t ws_size,
                              hipStream_t stream) {
    const float* x_obs = (const float*)d_in[IN_XOBS];
    const float* alpha = (const float*)d_in[IN_ALPHA];
    const int*   nom   = (const int*)d_in[IN_NOM];

    // workspace carve-out (float units, 256B aligned)
    float* wsf = (float*)d_ws;
    size_t off = 0;
    auto take = [&](size_t n) -> float* {
        float* p = wsf + off;
        off += (n + 63) & ~(size_t)63;
        return p;
    };
    float* P0   = take(4194304);   // B*32*64*64 max feature map
    float* P1   = take(4194304);
    float* z    = take(BATCH * 128);
    float* h0   = take(BATCH * 64);
    float* h1   = take(BATCH * 64);
    float* h2   = take(BATCH * 64);
    float* zl   = take(BATCH * 64);
    float* hm   = take(BATCH * 64);
    float* plog = take(BATCH * 1023);
    float* elog = take(BATCH * 1023);
    float* mis  = take(BATCH * T_STEPS);
    float* pp   = take(BATCH * T_STEPS);
    float* le   = take(BATCH * T_STEPS);
    float* spi  = take(BATCH * T_STEPS);
    float* sp   = take(BATCH * T_STEPS);
    float* se   = take(BATCH * T_STEPS);
    float* sa   = take(BATCH * T_STEPS);
    float* conf = take(BATCH * T_STEPS);
    int*   amin = (int*)take(BATCH * T_STEPS);
    int*   hard = (int*)take(BATCH * T_STEPS);

    auto cdiv = [](int a, int b) { return (a + b - 1) / b; };

    // ---- encoder: block 1 (1 -> 32, 128 -> 64) ----
    conv3x3_k<<<cdiv(BATCH * 32 * 64 * 64, 256), 256, 0, stream>>>(
        x_obs, BLK_W1(B0), BLK_B1(B0), P0, 1, 32, 128, 128, 2, 64, 64);
    gn_gelu_k<<<BATCH * 4, 256, 0, stream>>>(P0, BLK_G1(B0), BLK_BG1(B0), P1, 32, 64 * 64);
    conv3x3_k<<<cdiv(BATCH * 32 * 64 * 64, 256), 256, 0, stream>>>(
        P1, BLK_W2(B0), BLK_B2(B0), P0, 32, 32, 64, 64, 1, 64, 64);
    gn_gelu_k<<<BATCH * 4, 256, 0, stream>>>(P0, BLK_G2(B0), BLK_BG2(B0), P1, 32, 64 * 64);

    // ---- block 2 (32 -> 64, 64 -> 32) ----
    conv3x3_k<<<cdiv(BATCH * 64 * 32 * 32, 256), 256, 0, stream>>>(
        P1, BLK_W1(B1), BLK_B1(B1), P0, 32, 64, 64, 64, 2, 32, 32);
    gn_gelu_k<<<BATCH * 4, 256, 0, stream>>>(P0, BLK_G1(B1), BLK_BG1(B1), P1, 64, 32 * 32);
    conv3x3_k<<<cdiv(BATCH * 64 * 32 * 32, 256), 256, 0, stream>>>(
        P1, BLK_W2(B1), BLK_B2(B1), P0, 64, 64, 32, 32, 1, 32, 32);
    gn_gelu_k<<<BATCH * 4, 256, 0, stream>>>(P0, BLK_G2(B1), BLK_BG2(B1), P1, 64, 32 * 32);

    // ---- block 3 (64 -> 128, 32 -> 16) ----
    conv3x3_k<<<cdiv(BATCH * 128 * 16 * 16, 256), 256, 0, stream>>>(
        P1, BLK_W1(B2), BLK_B1(B2), P0, 64, 128, 32, 32, 2, 16, 16);
    gn_gelu_k<<<BATCH * 4, 256, 0, stream>>>(P0, BLK_G1(B2), BLK_BG1(B2), P1, 128, 16 * 16);
    conv3x3_k<<<cdiv(BATCH * 128 * 16 * 16, 256), 256, 0, stream>>>(
        P1, BLK_W2(B2), BLK_B2(B2), P0, 128, 128, 16, 16, 1, 16, 16);
    gn_gelu_k<<<BATCH * 4, 256, 0, stream>>>(P0, BLK_G2(B2), BLK_BG2(B2), P1, 128, 16 * 16);

    // ---- pool + MLP heads (LDS-staged WMMA GEMMs) ----
    avgpool_k<<<cdiv(BATCH * 128, 256), 256, 0, stream>>>(P1, z);

    dim3 b128(128, 1, 1);
    int g64  = 1;                 // N = 64  -> one 32x64 tile
    int g1023 = cdiv(1023, 64);   // 16 tiles
    int g1024 = cdiv(1024, 64);   // 16 tiles
    // proj: 128->64 gelu, 64->64 none
    gemm_wmma_k<128, 1><<<g64, b128, 0, stream>>>(z,  (const float*)d_in[PR_W0], (const float*)d_in[PR_B0], h0, 64);
    gemm_wmma_k< 64, 0><<<g64, b128, 0, stream>>>(h0, (const float*)d_in[PR_W1], (const float*)d_in[PR_B1], zl, 64);
    // phase: 64->64 gelu, 64->64 gelu, 64->1023 none
    gemm_wmma_k< 64, 1><<<g64, b128, 0, stream>>>(zl, (const float*)d_in[PH_W0], (const float*)d_in[PH_B0], h1, 64);
    gemm_wmma_k< 64, 1><<<g64, b128, 0, stream>>>(h1, (const float*)d_in[PH_W1], (const float*)d_in[PH_B1], h2, 64);
    gemm_wmma_k< 64, 0><<<g1023, b128, 0, stream>>>(h2, (const float*)d_in[PH_W2], (const float*)d_in[PH_B2], plog, 1023);
    // energy: same topology
    gemm_wmma_k< 64, 1><<<g64, b128, 0, stream>>>(zl, (const float*)d_in[EN_W0], (const float*)d_in[EN_B0], h1, 64);
    gemm_wmma_k< 64, 1><<<g64, b128, 0, stream>>>(h1, (const float*)d_in[EN_W1], (const float*)d_in[EN_B1], h2, 64);
    gemm_wmma_k< 64, 0><<<g1023, b128, 0, stream>>>(h2, (const float*)d_in[EN_W2], (const float*)d_in[EN_B2], elog, 1023);
    // mismatch: 64->64 gelu, 64->1024 sigmoid
    gemm_wmma_k< 64, 1><<<g64, b128, 0, stream>>>(zl, (const float*)d_in[MM_W0], (const float*)d_in[MM_B0], hm, 64);
    gemm_wmma_k< 64, 2><<<g1024, b128, 0, stream>>>(hm, (const float*)d_in[MM_W1], (const float*)d_in[MM_B1], mis, 1024);

    // ---- monotone heads ----
    monotone_k<<<1, BATCH, 0, stream>>>(plog, pp, 0);
    monotone_k<<<1, BATCH, 0, stream>>>(elog, le, 1);   // latent_energy = 1 - monotone

    // ---- fused assignment rows (writes 128 MB assignment once) ----
    float* assignment = (float*)d_out;
    assign_rows_k<<<BATCH * T_STEPS, 256, 0, stream>>>(
        pp, le, mis, alpha, assignment, spi, sp, se, sa, conf, amin);

    // ---- hard path + gathers + loss ----
    hard_cummax_k<<<1, BATCH, 0, stream>>>(amin, hard);

    float* outF      = (float*)d_out;
    float* out_phase = outF + (size_t)BATCH * T_STEPS * T_STEPS;        // 33554432
    float* out_alpha = out_phase + BATCH;
    float* out_loss  = out_alpha + BATCH;
    float* out_hard  = out_loss + 1;

    gather_out_k<<<1, BATCH, 0, stream>>>(nom, spi, sa, hard, out_phase, out_alpha, out_hard);
    loss_k<<<1, 256, 0, stream>>>(mis, se, sp, conf, out_loss);
}
